// NoisyTopkRouter_52561809768844
// MI455X (gfx1250) — matrix operand order repaired and checked by reading, not compiled, not run on hardware
//
#include <hip/hip_runtime.h>
#include <hip/hip_bf16.h>

#define BDIM 8
#define SDIM 4096
#define DDIM 2048
#define EDIM 64
#define NTOK (BDIM * SDIM) /* 32768 tokens */

typedef __attribute__((ext_vector_type(16))) __bf16 bf16x16;
typedef __attribute__((ext_vector_type(8)))  float  f32x8;

// -------- packed-B fragment addressing --------------------------------------
// One B fragment = 32x16 (KxN) bf16 tile in this code's contiguous-K order (A matches):
//   lane n      (n<16): halves h=0..15 hold B(k= h,      n)   i.e. K 0..15
//   lane 16+n         : halves h=0..15 hold B(k= 16+h,   n)   i.e. K 16..31
// Each lane's 16 halves are contiguous (32 bytes): one aligned load.
// Fragment id: [mat(2)][part hi/lo(2)][kstep(64)][ntile(4)] * 512 halves.
__device__ __host__ __forceinline__ size_t packb_off(int mat, int part, int kstep, int ntile) {
    return ((((size_t)(mat * 2 + part) * 64 + (size_t)kstep) * 4 + (size_t)ntile) << 9);
}

// -------- kernel 1: split f32 weights into hi/lo bf16, pre-swizzled ---------
__global__ __launch_bounds__(256)
void pack_weights_kernel(const float* __restrict__ w_route,
                         const float* __restrict__ w_noise,
                         __bf16* __restrict__ packB) {
    int idx = blockIdx.x * blockDim.x + threadIdx.x; // over 2*E*D
    if (idx >= 2 * EDIM * DDIM) return;
    int mat = idx / (EDIM * DDIM);
    int rem = idx - mat * (EDIM * DDIM);
    int e = rem / DDIM;
    int k = rem - e * DDIM;

    float w = (mat ? w_noise : w_route)[(size_t)e * DDIM + k];
    __bf16 hi = (__bf16)w;                     // RNE
    __bf16 lo = (__bf16)(w - (float)hi);       // residual

    int kstep = k >> 5;
    int kin   = k & 31;
    int ntile = e >> 4;
    int lane  = (e & 15) + ((kin >> 4) << 4);  // K>=16 lives in lanes 16..31
    int h     = kin & 15;

    packB[packb_off(mat, 0, kstep, ntile) + (size_t)lane * 16 + h] = hi;
    packB[packb_off(mat, 1, kstep, ntile) + (size_t)lane * 16 + h] = lo;
}

// -------- kernel 2: dual GEMM [32768,2048]x[2048,64] via bf16x3 WMMA --------
// Block = 256 threads = 8 waves. Each wave: 2 M-tiles (32 tokens) x 4 N-tiles
// x 2 matrices. 48 v_wmma per K-step, 64 K-steps.
__global__ __launch_bounds__(256)
void router_gemm_kernel(const float* __restrict__ A,       // [NTOK, D]
                        const __bf16* __restrict__ packB,
                        float* __restrict__ outR,          // [NTOK, E] raw route logits
                        float* __restrict__ outN) {        // [NTOK, E] raw noise logits
    const int lane = threadIdx.x & 31;
    const int wave = threadIdx.x >> 5;
    const int m0   = blockIdx.x * 256 + wave * 32; // first token of this wave

    f32x8 acc[2][2][4]; // [mtile][mat][ntile]
    #pragma unroll
    for (int mt = 0; mt < 2; ++mt)
        #pragma unroll
        for (int mat = 0; mat < 2; ++mat)
            #pragma unroll
            for (int t = 0; t < 4; ++t)
                acc[mt][mat][t] = {};

    // A-fragment k order used here (contiguous, 16x32 MxK):
    //  lanes 0-15 : M=lane,    halves 0..15 -> K 0..15
    //  lanes 16-31: M=lane-16, halves 0..15 -> K 16..31
    const int laneM = lane & 15;
    const int kb    = (lane >> 4) << 4; // 0 or 16
    const float* a0 = A + (size_t)(m0 + laneM) * DDIM + kb;
    const float* a1 = A + (size_t)(m0 + 16 + laneM) * DDIM + kb;

    for (int ks = 0; ks < DDIM / 32; ++ks) {
        const int k0 = ks * 32;
        bf16x16 ahi[2], alo[2];
        const float* ap[2] = { a0 + k0, a1 + k0 };
        #pragma unroll
        for (int mt = 0; mt < 2; ++mt) {
            #pragma unroll
            for (int j = 0; j < 8; ++j) {
                float x0 = ap[mt][j];
                float x1 = ap[mt][8 + j];
                __bf16 h0 = (__bf16)x0;
                __bf16 h1 = (__bf16)x1;
                ahi[mt][j]     = h0;
                alo[mt][j]     = (__bf16)(x0 - (float)h0);
                ahi[mt][8 + j] = h1;
                alo[mt][8 + j] = (__bf16)(x1 - (float)h1);
            }
        }
        #pragma unroll
        for (int mat = 0; mat < 2; ++mat) {
            #pragma unroll
            for (int t = 0; t < 4; ++t) {
                const __bf16* ph = packB + packb_off(mat, 0, ks, t) + (size_t)lane * 16;
                const __bf16* pl = packB + packb_off(mat, 1, ks, t) + (size_t)lane * 16;
                bf16x16 bh = *(const bf16x16*)ph;
                bf16x16 bl = *(const bf16x16*)pl;
                #pragma unroll
                for (int mt = 0; mt < 2; ++mt) {
                    acc[mt][mat][t] = __builtin_amdgcn_wmma_f32_16x16x32_bf16(
                        false, ahi[mt], false, bh, (short)0, acc[mt][mat][t], false, false);
                    acc[mt][mat][t] = __builtin_amdgcn_wmma_f32_16x16x32_bf16(
                        false, alo[mt], false, bh, (short)0, acc[mt][mat][t], false, false);
                    acc[mt][mat][t] = __builtin_amdgcn_wmma_f32_16x16x32_bf16(
                        false, ahi[mt], false, bl, (short)0, acc[mt][mat][t], false, false);
                }
            }
        }
    }

    // C/D layout: VGPR r -> lanes 0-15: M=r, N=lane; lanes 16-31: M=8+r, N=lane-16
    const int tokHalf = (lane < 16) ? 0 : 8;
    const int nlo     = lane & 15;
    #pragma unroll
    for (int mt = 0; mt < 2; ++mt) {
        #pragma unroll
        for (int mat = 0; mat < 2; ++mat) {
            float* outp = mat ? outN : outR;
            #pragma unroll
            for (int t = 0; t < 4; ++t) {
                const int expert = t * 16 + nlo;
                #pragma unroll
                for (int r = 0; r < 8; ++r) {
                    const int token = m0 + mt * 16 + tokHalf + r;
                    outp[(size_t)token * EDIM + expert] = acc[mt][mat][t][r];
                }
            }
        }
    }
}

// -------- kernel 3: per-token epilogue --------------------------------------
__global__ __launch_bounds__(256)
void router_epilogue_kernel(const float* __restrict__ logitsS,
                            const float* __restrict__ nlogS,
                            const float* __restrict__ b_route,
                            const float* __restrict__ b_noise,
                            const float* __restrict__ noise,
                            float* __restrict__ outRouter,
                            float* __restrict__ outIdx,
                            float* __restrict__ outGate) {
    const int t = blockIdx.x * blockDim.x + threadIdx.x;
    if (t >= NTOK) return;
    const size_t base = (size_t)t * EDIM;

    float noisy[EDIM];
    float v1 = -3.402823466e38f, v2 = -3.402823466e38f;
    int i1 = 0, i2 = 0;
    #pragma unroll
    for (int e = 0; e < EDIM; ++e) {
        float l  = logitsS[base + e] + b_route[e];
        float nl = nlogS[base + e]   + b_noise[e];
        float sp = (nl > 20.0f) ? nl : log1pf(__expf(nl)); // softplus
        float v  = l + noise[base + e] * sp;
        noisy[e] = v;
        if (v > v1)      { v2 = v1; i2 = i1; v1 = v; i1 = e; }
        else if (v > v2) { v2 = v;  i2 = e; }
    }
    float sum = 0.0f;
    #pragma unroll
    for (int e = 0; e < EDIM; ++e) {
        float g = __expf(noisy[e] - v1);
        noisy[e] = g;
        sum += g;
    }
    const float inv   = 1.0f / sum;
    const float e2    = __expf(v2 - v1);
    const float s2inv = 1.0f / (1.0f + e2);
    #pragma unroll
    for (int e = 0; e < EDIM; ++e) {
        outGate[base + e] = noisy[e] * inv;
        float r = (e == i1) ? s2inv : ((e == i2) ? e2 * s2inv : 0.0f);
        outRouter[base + e] = r;
    }
    outIdx[(size_t)t * 2 + 0] = (float)i1;
    outIdx[(size_t)t * 2 + 1] = (float)i2;
}

// -------- host launch -------------------------------------------------------
extern "C" void kernel_launch(void* const* d_in, const int* in_sizes, int n_in,
                              void* d_out, int out_size, void* d_ws, size_t ws_size,
                              hipStream_t stream) {
    const float* mh    = (const float*)d_in[0]; // [B,S,D]
    const float* wr    = (const float*)d_in[1]; // [E,D]
    const float* br    = (const float*)d_in[2]; // [E]
    const float* wn    = (const float*)d_in[3]; // [E,D]
    const float* bn    = (const float*)d_in[4]; // [E]
    const float* noise = (const float*)d_in[5]; // [B,S,E]
    // d_in[6] = top_k == 2 (compile-time constant here)

    __bf16* packB  = (__bf16*)d_ws;                                   // 1 MB
    float* logitsS = (float*)((char*)d_ws +
                     (size_t)(2 * 2 * EDIM * DDIM) * sizeof(__bf16)); // 8 MB
    float* nlogS   = logitsS + (size_t)NTOK * EDIM;                   // 8 MB

    float* outRouter = (float*)d_out;                      // [NTOK, E]
    float* outIdx    = outRouter + (size_t)NTOK * EDIM;    // [NTOK, 2]
    float* outGate   = outIdx + (size_t)NTOK * 2;          // [NTOK, E]

    pack_weights_kernel<<<(2 * EDIM * DDIM + 255) / 256, 256, 0, stream>>>(wr, wn, packB);
    router_gemm_kernel<<<NTOK / 256, 256, 0, stream>>>(mh, packB, logitsS, nlogS);
    router_epilogue_kernel<<<NTOK / 256, 256, 0, stream>>>(logitsS, nlogS, br, bn, noise,
                                                           outRouter, outIdx, outGate);
}
